// FixedPointViT_82343112999079
// MI455X (gfx1250) — compile-verified, matrix operand order checked by reading, not run on hardware
//
#include <hip/hip_runtime.h>
#include <hip/hip_bf16.h>

typedef __attribute__((ext_vector_type(16))) _Float16 v16h;
typedef __attribute__((ext_vector_type(8)))  _Float16 v8h;
typedef __attribute__((ext_vector_type(4)))  _Float16 v4h;
typedef __attribute__((ext_vector_type(8)))  float    v8f;
typedef int v4i __attribute__((vector_size(16)));   // matches builtin's int4

#define BATCH   16
#define CCH     3
#define IMGW    224
#define PSZ     16
#define DIMF    768
#define HEADS   12
#define DHEAD   64
#define MLPD    3072
#define NCLS    1000
#define GRIDP   14
#define NPATCH  196
#define TSEQ    197
#define TP      208      // 13*16, padded sequence length
#define PATCHD  768      // C*P*P

// Async global->LDS copy path (CDNA5): use if the toolchain exposes it.
#if defined(__AMDGCN__) && __has_builtin(__builtin_amdgcn_global_load_async_to_lds_b128) && __has_builtin(__builtin_amdgcn_s_wait_asynccnt)
#define ASYNC_LDS 1
typedef __attribute__((address_space(1))) v4i* gv4i_p;
typedef __attribute__((address_space(3))) v4i* lv4i_p;
#else
#define ASYNC_LDS 0
#endif

__device__ __forceinline__ float fxp(float x) {
    x = fminf(fmaxf(x, -1.0f), 1.0f);
    return truncf(x * 64.0f) * 0.015625f;
}
__device__ __forceinline__ float gelu_exact(float x) {
    return 0.5f * x * (1.0f + erff(x * 0.70710678118654752f));
}
__device__ __forceinline__ float block_sum(float v, float* sbuf) {
    int tid = threadIdx.x;
    sbuf[tid] = v; __syncthreads();
#pragma unroll
    for (int s = 128; s > 0; s >>= 1) {
        if (tid < s) sbuf[tid] += sbuf[tid + s];
        __syncthreads();
    }
    float r = sbuf[0]; __syncthreads();
    return r;
}

// ---------------------------------------------------------------------------
// fp32 [K][N] -> f16 [N][K] transposed conversion (LDS-tiled, 32x32 tiles)
// Weight GEMMs then consume B row-major-in-K = pure b128 copy staging.
// ---------------------------------------------------------------------------
__global__ __launch_bounds__(256)
void k_cvt16t(const float* __restrict__ src, _Float16* __restrict__ dst, int K, int N)
{
    __shared__ _Float16 tile[32][33];
    const int k0 = blockIdx.x * 32;
    const int n0 = blockIdx.y * 32;
    const int tx = threadIdx.x & 31, ty = threadIdx.x >> 5;   // ty in [0,8)
#pragma unroll
    for (int j = 0; j < 4; ++j) {
        int k = k0 + ty + 8 * j, n = n0 + tx;
        float v = (k < K && n < N) ? src[(long)k * N + n] : 0.0f;
        tile[ty + 8 * j][tx] = (_Float16)v;
    }
    __syncthreads();
#pragma unroll
    for (int j = 0; j < 4; ++j) {
        int n = n0 + ty + 8 * j, k = k0 + tx;
        if (n < N && k < K) dst[(long)n * K + k] = tile[tx][ty + 8 * j];
    }
}

// ---------------------------------------------------------------------------
// WMMA GEMM: C[M,N] = epilogue(A[M,K](f32) @ B(TB))
//   BT=true : B row-major [N,K]  (f16 pre-transposed weights, or Q K^T f32)
//   BT=false: B row-major [K,N]  (attention V, f32)
// Block tile 128x64, K-step 32, 8 waves; each wave -> 32x32 (2x2 WMMA tiles).
// LDS tiles stored fragment-friendly: As[M][K], Bs[N][K]; each 16x32 fragment
// is two contiguous 16-byte runs per lane -> 2x ds_load_b128.
// f16 B tiles staged with GLOBAL_LOAD_ASYNC_TO_LDS_B128 when available.
// ---------------------------------------------------------------------------
enum { EP_NONE = 0, EP_BIAS, EP_FXP, EP_SCALE_FXP, EP_BIAS_FXP_GELU_FXP, EP_BIAS_FXP_RES_FXP };

template <typename TB, bool BT, int EPI>
__global__ __launch_bounds__(256)
void k_gemm(const float* A, long lda,
            const TB* Bm, long ldb,
            float* C, long ldc,
            int M, int N, int K,
            const float* bias,
            const float* res,
            float scale, int H,
            long sAb, long sAh, long sBb, long sBh, long sCb, long sCh)
{
    __shared__ __align__(16) _Float16 As[128][40];  // [M][K], row = 80B (5 x v8h)
    __shared__ __align__(16) _Float16 Bs[64][40];   // [N][K]

    const int z  = blockIdx.z;
    const int bb = z / H;
    const int hh = z % H;
    A  += (long)bb * sAb + (long)hh * sAh;
    Bm += (long)bb * sBb + (long)hh * sBh;
    C  += (long)bb * sCb + (long)hh * sCh;
    const float* resp = res;
    if (EPI == EP_BIAS_FXP_RES_FXP) resp += (long)bb * sCb + (long)hh * sCh;

    const int bm = blockIdx.y * 128;
    const int bn = blockIdx.x * 64;
    const int tid  = threadIdx.x;
    const int lane = tid & 31;
    const int w    = tid >> 5;
    const int wm = (w & 3) * 32;    // wave row offset (M)
    const int wn = (w >> 2) * 32;   // wave col offset (N)
    const int g = lane >> 4;        // half-wave select
    const int m = lane & 15;

    const bool fullM = (bm + 128 <= M);
    const bool fullN = (bn + 64 <= N);

    v8f acc[2][2] = {};

    for (int k0 = 0; k0 < K; k0 += 32) {
        const bool fullK = (k0 + 32 <= K);

        // ---- prefetch next K-step tiles (global_prefetch_b8, near scope) ----
        if (k0 + 32 < K) {
            if (fullM)
                __builtin_prefetch(A + (long)(bm + (tid >> 1)) * lda + (k0 + 32) + (tid & 1) * 16, 0, 3);
            if (fullN) {
                if (BT) { if (tid < 64) __builtin_prefetch(Bm + (long)(bn + tid) * ldb + (k0 + 32), 0, 3); }
                else    { if (tid < 32) __builtin_prefetch(Bm + (long)(k0 + 32 + tid) * ldb + bn, 0, 3); }
            }
        }

        // ---------------- stage A tile (128x32) ----------------
        if (fullM && fullK) {
#pragma unroll
            for (int it = 0; it < 4; ++it) {
                int idx = tid + it * 256;            // [0,1024)
                int r = idx >> 3, c4 = (idx & 7) * 4;
                float4 f = *(const float4*)(A + (long)(bm + r) * lda + (k0 + c4));
                v4h hv;
                hv[0] = (_Float16)f.x; hv[1] = (_Float16)f.y;
                hv[2] = (_Float16)f.z; hv[3] = (_Float16)f.w;
                *(v4h*)&As[r][c4] = hv;
            }
        } else {
            for (int i = tid; i < 128 * 32; i += 256) {
                int r = i >> 5, c = i & 31;
                int gr = bm + r, gc = k0 + c;
                float v = (gr < M && gc < K) ? A[(long)gr * lda + gc] : 0.0f;
                As[r][c] = (_Float16)v;
            }
        }

        // ---------------- stage B tile as [N][K] (64x32) ----------------
        if (fullN && fullK) {
            if (BT) {
                if (sizeof(TB) == 2) {
                    // pre-transposed f16 weights: pure 16B copy per lane
                    int r = tid >> 2, c8 = (tid & 3) * 8;
                    const _Float16* gp = (const _Float16*)Bm + (long)(bn + r) * ldb + (k0 + c8);
                    _Float16* lp = &Bs[r][c8];
#if ASYNC_LDS
                    __builtin_amdgcn_global_load_async_to_lds_b128(
                        (gv4i_p)(void*)gp, (lv4i_p)(void*)lp, 0, 0);
#else
                    *(v8h*)lp = *(const v8h*)gp;
#endif
                } else {
                    // f32 [N,K] (Q K^T): vector load + convert
#pragma unroll
                    for (int it = 0; it < 2; ++it) {
                        int idx = tid + it * 256;        // [0,512)
                        int r = idx >> 3, c4 = (idx & 7) * 4;
                        float4 f = *(const float4*)((const float*)Bm + (long)(bn + r) * ldb + (k0 + c4));
                        v4h hv;
                        hv[0] = (_Float16)f.x; hv[1] = (_Float16)f.y;
                        hv[2] = (_Float16)f.z; hv[3] = (_Float16)f.w;
                        *(v4h*)&Bs[r][c4] = hv;
                    }
                }
            } else {
                // f32 [K,N] (attention V): transpose while staging
#pragma unroll
                for (int it = 0; it < 2; ++it) {
                    int idx = tid + it * 256;
                    int n = idx >> 3, c4 = (idx & 7) * 4;
                    v4h hv;
#pragma unroll
                    for (int j = 0; j < 4; ++j)
                        hv[j] = (_Float16)(float)Bm[(long)(k0 + c4 + j) * ldb + (bn + n)];
                    *(v4h*)&Bs[n][c4] = hv;
                }
            }
        } else {
            for (int i = tid; i < 64 * 32; i += 256) {
                int r = i >> 5, c = i & 31;       // r = N index, c = K index
                int gn = bn + r, gk = k0 + c;
                bool ok = (gn < N && gk < K);
                float v;
                if (BT) v = ok ? (float)Bm[(long)gn * ldb + gk] : 0.0f;
                else    v = ok ? (float)Bm[(long)gk * ldb + gn] : 0.0f;
                Bs[r][c] = (_Float16)v;
            }
        }
#if ASYNC_LDS
        if (BT && sizeof(TB) == 2) __builtin_amdgcn_s_wait_asynccnt(0);
#endif
        __syncthreads();

        // ---- fragments: two contiguous v8h runs per lane -> ds_load_b128 ----
        const v8h* Ar0 = (const v8h*)&As[wm + m][0];
        const v8h* Ar1 = (const v8h*)&As[wm + 16 + m][0];
        const v8h* Br0 = (const v8h*)&Bs[wn + m][0];
        const v8h* Br1 = (const v8h*)&Bs[wn + 16 + m][0];
        v16h a0 = __builtin_shufflevector(Ar0[g], Ar0[2 + g],
                  0,1,2,3,4,5,6,7,8,9,10,11,12,13,14,15);
        v16h a1 = __builtin_shufflevector(Ar1[g], Ar1[2 + g],
                  0,1,2,3,4,5,6,7,8,9,10,11,12,13,14,15);
        v16h b0 = __builtin_shufflevector(Br0[g], Br0[2 + g],
                  0,1,2,3,4,5,6,7,8,9,10,11,12,13,14,15);
        v16h b1 = __builtin_shufflevector(Br1[g], Br1[2 + g],
                  0,1,2,3,4,5,6,7,8,9,10,11,12,13,14,15);

        acc[0][0] = __builtin_amdgcn_wmma_f32_16x16x32_f16(false, a0, false, b0, (short)0, acc[0][0], false, false);
        acc[0][1] = __builtin_amdgcn_wmma_f32_16x16x32_f16(false, a0, false, b1, (short)0, acc[0][1], false, false);
        acc[1][0] = __builtin_amdgcn_wmma_f32_16x16x32_f16(false, a1, false, b0, (short)0, acc[1][0], false, false);
        acc[1][1] = __builtin_amdgcn_wmma_f32_16x16x32_f16(false, a1, false, b1, (short)0, acc[1][1], false, false);
        __syncthreads();
    }

    // ---- epilogue + store (C/D layout: VGPR v -> rows v and v+8) ----
#pragma unroll
    for (int ti = 0; ti < 2; ++ti) {
#pragma unroll
        for (int v2 = 0; v2 < 8; ++v2) {
            int r = bm + wm + ti * 16 + 8 * g + v2;
            if (r < M) {
#pragma unroll
                for (int tj = 0; tj < 2; ++tj) {
                    int c = bn + wn + tj * 16 + m;
                    if (c < N) {
                        float a = acc[ti][tj][v2];
                        float v = a;
                        if      (EPI == EP_BIAS)      v = a + bias[c];
                        else if (EPI == EP_FXP)       v = fxp(a);
                        else if (EPI == EP_SCALE_FXP) v = fxp(a * scale);
                        else if (EPI == EP_BIAS_FXP_GELU_FXP) { v = fxp(a + bias[c]); v = fxp(gelu_exact(v)); }
                        else if (EPI == EP_BIAS_FXP_RES_FXP)  { v = fxp(a + bias[c]); v = fxp(v + resp[(long)r * ldc + c]); }
                        C[(long)r * ldc + c] = v;
                    }
                }
            }
        }
    }
}

// ---------------------------------------------------------------------------
// Patch extraction: fxp(img) -> patch rows -> LayerNorm(patch_ln) -> Xp
// ---------------------------------------------------------------------------
__global__ __launch_bounds__(256)
void k_patchify(const float* __restrict__ img,
                const float* __restrict__ pg, const float* __restrict__ pb,
                float* __restrict__ Xp)
{
    __shared__ float sbuf[256];
    const int p = blockIdx.x, b = blockIdx.y;
    const int gy = p / GRIDP, gx = p % GRIDP;
    const int tid = threadIdx.x;
    float v[3];
#pragma unroll
    for (int k = 0; k < 3; ++k) {
        int j  = tid + k * 256;               // j = (py*16+px)*3 + c
        int py = j / 48;
        int px = (j / 3) % 16;
        int c  = j % 3;
        long src = (((long)(b * CCH + c) * IMGW) + (gy * PSZ + py)) * IMGW + (gx * PSZ + px);
        v[k] = fxp(img[src]);
    }
    float s  = block_sum(v[0] + v[1] + v[2], sbuf);
    float mu = s * (1.0f / 768.0f);
    float d0 = v[0] - mu, d1 = v[1] - mu, d2 = v[2] - mu;
    float sq = block_sum(d0 * d0 + d1 * d1 + d2 * d2, sbuf);
    float rs = rsqrtf(sq * (1.0f / 768.0f) + 1e-5f);
    long row = ((long)b * NPATCH + p) * PATCHD;
    Xp[row + tid]       = d0 * rs * pg[tid]       + pb[tid];
    Xp[row + tid + 256] = d1 * rs * pg[tid + 256] + pb[tid + 256];
    Xp[row + tid + 512] = d2 * rs * pg[tid + 512] + pb[tid + 512];
}

// ---------------------------------------------------------------------------
// Embed: x[b,t] = fxp( (t==0 ? cls : fxp(LN(emb[b,t-1]))) + pos[t] ), pad->0
// ---------------------------------------------------------------------------
__global__ __launch_bounds__(256)
void k_embed(const float* __restrict__ Emb,
             const float* __restrict__ eg, const float* __restrict__ eb,
             const float* __restrict__ cls, const float* __restrict__ pos,
             float* __restrict__ x)
{
    __shared__ float sbuf[256];
    const int t = blockIdx.x, b = blockIdx.y;
    const int tid = threadIdx.x;
    float* yp = x + ((long)b * TP + t) * DIMF;
    if (t >= TSEQ) {
        yp[tid] = 0.0f; yp[tid + 256] = 0.0f; yp[tid + 512] = 0.0f;
        return;
    }
    if (t == 0) {
#pragma unroll
        for (int k = 0; k < 3; ++k) {
            int d = tid + k * 256;
            yp[d] = fxp(cls[d] + pos[d]);
        }
        return;
    }
    const float* ep = Emb + ((long)b * NPATCH + (t - 1)) * DIMF;
    float v0 = ep[tid], v1 = ep[tid + 256], v2 = ep[tid + 512];
    float s  = block_sum(v0 + v1 + v2, sbuf);
    float mu = s * (1.0f / 768.0f);
    float d0 = v0 - mu, d1 = v1 - mu, d2 = v2 - mu;
    float sq = block_sum(d0 * d0 + d1 * d1 + d2 * d2, sbuf);
    float rs = rsqrtf(sq * (1.0f / 768.0f) + 1e-5f);
    const float* pp = pos + (long)t * DIMF;
#pragma unroll
    for (int k = 0; k < 3; ++k) {
        int d = tid + k * 256;
        float dd = (k == 0 ? d0 : (k == 1 ? d1 : d2));
        float v  = fxp(dd * rs * eg[d] + eb[d]);
        yp[d] = fxp(v + pp[d]);
    }
}

// ---------------------------------------------------------------------------
// LayerNorm over DIMF, optional fxp.  One block per row.
// ---------------------------------------------------------------------------
template <bool DOFXP>
__global__ __launch_bounds__(256)
void k_ln(const float* __restrict__ X, long inStride,
          float* __restrict__ Y,
          const float* __restrict__ g, const float* __restrict__ b)
{
    __shared__ float sbuf[256];
    const long row = blockIdx.x;
    const float* xp = X + row * inStride;
    float* yp = Y + row * (long)DIMF;
    const int tid = threadIdx.x;
    float v0 = xp[tid], v1 = xp[tid + 256], v2 = xp[tid + 512];
    float s  = block_sum(v0 + v1 + v2, sbuf);
    float mu = s * (1.0f / 768.0f);
    float d0 = v0 - mu, d1 = v1 - mu, d2 = v2 - mu;
    float sq = block_sum(d0 * d0 + d1 * d1 + d2 * d2, sbuf);
    float rs = rsqrtf(sq * (1.0f / 768.0f) + 1e-5f);
#pragma unroll
    for (int k = 0; k < 3; ++k) {
        int d = tid + k * 256;
        float dd = (k == 0 ? d0 : (k == 1 ? d1 : d2));
        float o  = dd * rs * g[d] + b[d];
        if (DOFXP) o = fxp(o);
        yp[d] = o;
    }
}

// ---------------------------------------------------------------------------
// Row softmax over first TSEQ cols of a TP-wide row; fxp output, pad cols->0
// ---------------------------------------------------------------------------
__global__ __launch_bounds__(256)
void k_softmax(float* __restrict__ S)
{
    __shared__ float sbuf[256];
    const long row = blockIdx.x;
    float* p = S + row * (long)TP;
    const int tid = threadIdx.x;
    float v = (tid < TSEQ) ? p[tid] : -1e30f;
    sbuf[tid] = v; __syncthreads();
#pragma unroll
    for (int s = 128; s > 0; s >>= 1) {
        if (tid < s) sbuf[tid] = fmaxf(sbuf[tid], sbuf[tid + s]);
        __syncthreads();
    }
    float mx = sbuf[0]; __syncthreads();
    float e = (tid < TSEQ) ? expf(v - mx) : 0.0f;
    float sum = block_sum(e, sbuf);
    float inv = 1.0f / sum;
    if (tid < TP) p[tid] = (tid < TSEQ) ? fxp(e * inv) : 0.0f;
}

// ---------------------------------------------------------------------------
extern "C" void kernel_launch(void* const* d_in, const int* in_sizes, int n_in,
                              void* d_out, int out_size, void* d_ws, size_t ws_size,
                              hipStream_t stream)
{
    const float* img        = (const float*)d_in[0];
    const float* patch_ln_g = (const float*)d_in[1];
    const float* patch_ln_b = (const float*)d_in[2];
    const float* w_patch    = (const float*)d_in[3];
    const float* b_patch    = (const float*)d_in[4];
    const float* emb_ln_g   = (const float*)d_in[5];
    const float* emb_ln_b   = (const float*)d_in[6];
    const float* pos_emb    = (const float*)d_in[7];
    const float* cls_token  = (const float*)d_in[8];
    const float* ln1_g      = (const float*)d_in[9];
    const float* ln1_b      = (const float*)d_in[10];
    const float* w_qkv      = (const float*)d_in[11];
    const float* w_proj     = (const float*)d_in[12];
    const float* b_proj     = (const float*)d_in[13];
    const float* ln2_g      = (const float*)d_in[14];
    const float* ln2_b      = (const float*)d_in[15];
    const float* w_ff1      = (const float*)d_in[16];
    const float* b_ff1      = (const float*)d_in[17];
    const float* w_ff2      = (const float*)d_in[18];
    const float* b_ff2      = (const float*)d_in[19];
    const float* head_ln_g  = (const float*)d_in[20];
    const float* head_ln_b  = (const float*)d_in[21];
    const float* w_head     = (const float*)d_in[22];
    const float* b_head     = (const float*)d_in[23];
    float* out = (float*)d_out;

    // ---- workspace layout ----
    char* base = (char*)d_ws;
    size_t o = 0;
    auto take = [&](size_t nbytes) -> void* {
        void* p = (void*)(base + o);
        o += ((nbytes + 255) / 256) * 256;
        return p;
    };
    const long Mrows = (long)BATCH * TP;                       // 3328
    float* x      = (float*)take((size_t)Mrows * DIMF * 4);
    float* h      = (float*)take((size_t)Mrows * DIMF * 4);
    float* qkv    = (float*)take((size_t)Mrows * 3 * DIMF * 4);
    float* S      = (float*)take((size_t)BATCH * HEADS * TP * TP * 4);
    float* att    = (float*)take((size_t)Mrows * DIMF * 4);
    float* ff     = (float*)take((size_t)Mrows * MLPD * 4);
    float* pooled = (float*)take((size_t)BATCH * DIMF * 4);
    _Float16* wf16 = (_Float16*)take((size_t)DIMF * MLPD * 2); // f16 [N][K] weight scratch
    float* Xp  = ff;   // patch-stage scratch aliases MLP buffer (written later)
    float* Emb = S;    // embed-stage scratch aliases score buffer

    // transpose-convert a [K][N] fp32 weight into wf16 as [N][K] f16
    auto cvtT = [&](const float* src, int K, int N) {
        k_cvt16t<<<dim3((K + 31) / 32, (N + 31) / 32), 256, 0, stream>>>(src, wf16, K, N);
    };

    // ---- patch embedding ----
    k_patchify<<<dim3(NPATCH, BATCH), 256, 0, stream>>>(img, patch_ln_g, patch_ln_b, Xp);
    cvtT(w_patch, DIMF, DIMF);
    k_gemm<_Float16, true, EP_BIAS><<<dim3(DIMF / 64, (BATCH * NPATCH + 127) / 128, 1), 256, 0, stream>>>(
        Xp, DIMF, wf16, DIMF, Emb, DIMF,
        BATCH * NPATCH, DIMF, DIMF,
        b_patch, nullptr, 1.0f, 1, 0, 0, 0, 0, 0, 0);
    k_embed<<<dim3(TP, BATCH), 256, 0, stream>>>(Emb, emb_ln_g, emb_ln_b, cls_token, pos_emb, x);

    // ---- transformer layers ----
    for (int i = 0; i < 12; ++i) {
        const float* l1g = ln1_g  + (size_t)i * DIMF;
        const float* l1b = ln1_b  + (size_t)i * DIMF;
        const float* wqk = w_qkv  + (size_t)i * DIMF * 3 * DIMF;
        const float* wpr = w_proj + (size_t)i * DIMF * DIMF;
        const float* bpr = b_proj + (size_t)i * DIMF;
        const float* l2g = ln2_g  + (size_t)i * DIMF;
        const float* l2b = ln2_b  + (size_t)i * DIMF;
        const float* wf1 = w_ff1  + (size_t)i * DIMF * MLPD;
        const float* bf1 = b_ff1  + (size_t)i * MLPD;
        const float* wf2 = w_ff2  + (size_t)i * MLPD * DIMF;
        const float* bf2 = b_ff2  + (size_t)i * DIMF;

        // h = fxp(LN1(x))
        k_ln<true><<<(int)Mrows, 256, 0, stream>>>(x, DIMF, h, l1g, l1b);
        // qkv = fxp(h @ w_qkv)
        cvtT(wqk, DIMF, 3 * DIMF);
        k_gemm<_Float16, true, EP_FXP><<<dim3(3 * DIMF / 64, (int)(Mrows / 128), 1), 256, 0, stream>>>(
            h, DIMF, wf16, DIMF, qkv, 3 * DIMF,
            (int)Mrows, 3 * DIMF, DIMF,
            nullptr, nullptr, 1.0f, 1, 0, 0, 0, 0, 0, 0);
        // S[b,h] = fxp( (Q K^T) / 8 )  (batched b*h, B=[N,K] f32)
        k_gemm<float, true, EP_SCALE_FXP><<<dim3((TP + 63) / 64, (TP + 127) / 128, BATCH * HEADS), 256, 0, stream>>>(
            qkv, 3 * DIMF, qkv + DIMF, 3 * DIMF, S, TP,
            TP, TP, DHEAD,
            nullptr, nullptr, 0.125f, HEADS,
            (long)TP * 3 * DIMF, DHEAD,
            (long)TP * 3 * DIMF, DHEAD,
            (long)HEADS * TP * TP, (long)TP * TP);
        // attn = fxp(softmax(S))  (mask padded cols -> 0)
        k_softmax<<<BATCH * HEADS * TP, 256, 0, stream>>>(S);
        // att[b, :, h*64:(h+1)*64] = fxp(attn @ V)   (B=[K,N] f32)
        k_gemm<float, false, EP_FXP><<<dim3(1, (TP + 127) / 128, BATCH * HEADS), 256, 0, stream>>>(
            S, TP, qkv + 2 * DIMF, 3 * DIMF, att, DIMF,
            TP, DHEAD, TP,
            nullptr, nullptr, 1.0f, HEADS,
            (long)HEADS * TP * TP, (long)TP * TP,
            (long)TP * 3 * DIMF, DHEAD,
            (long)TP * DIMF, DHEAD);
        // x = fxp( fxp(att @ w_proj + b_proj) + x )
        cvtT(wpr, DIMF, DIMF);
        k_gemm<_Float16, true, EP_BIAS_FXP_RES_FXP><<<dim3(DIMF / 64, (int)(Mrows / 128), 1), 256, 0, stream>>>(
            att, DIMF, wf16, DIMF, x, DIMF,
            (int)Mrows, DIMF, DIMF,
            bpr, x, 1.0f, 1, 0, 0, 0, 0, 0, 0);
        // h = fxp(LN2(x))
        k_ln<true><<<(int)Mrows, 256, 0, stream>>>(x, DIMF, h, l2g, l2b);
        // ff = fxp(gelu(fxp(h @ w_ff1 + b_ff1)))
        cvtT(wf1, DIMF, MLPD);
        k_gemm<_Float16, true, EP_BIAS_FXP_GELU_FXP><<<dim3(MLPD / 64, (int)(Mrows / 128), 1), 256, 0, stream>>>(
            h, DIMF, wf16, DIMF, ff, MLPD,
            (int)Mrows, MLPD, DIMF,
            bf1, nullptr, 1.0f, 1, 0, 0, 0, 0, 0, 0);
        // x = fxp( fxp(ff @ w_ff2 + b_ff2) + x )
        cvtT(wf2, MLPD, DIMF);
        k_gemm<_Float16, true, EP_BIAS_FXP_RES_FXP><<<dim3(DIMF / 64, (int)(Mrows / 128), 1), 256, 0, stream>>>(
            ff, MLPD, wf16, MLPD, x, DIMF,
            (int)Mrows, DIMF, MLPD,
            bf2, x, 1.0f, 1, 0, 0, 0, 0, 0, 0);
    }

    // ---- head: pooled = LN(x[:,0]);  out = pooled @ w_head + b_head ----
    k_ln<false><<<BATCH, 256, 0, stream>>>(x, (long)TP * DIMF, pooled, head_ln_g, head_ln_b);
    cvtT(w_head, DIMF, NCLS);
    k_gemm<_Float16, true, EP_BIAS><<<dim3((NCLS + 63) / 64, 1, 1), 256, 0, stream>>>(
        pooled, DIMF, wf16, DIMF, out, NCLS,
        BATCH, NCLS, DIMF,
        b_head, nullptr, 1.0f, 1, 0, 0, 0, 0, 0, 0);
}
